// MSABlock_24412594111331
// MI455X (gfx1250) — matrix operand order repaired and checked by reading, not compiled, not run on hardware
//
#include <hip/hip_runtime.h>
#include <math.h>

// ---------------------------------------------------------------------------
// MI455X (gfx1250) Evoformer block. wave32, WMMA f16->f32 (16x16x32),
// async global->LDS tile staging (ASYNCcnt), fp16 operand pipeline.
// ---------------------------------------------------------------------------

typedef __attribute__((ext_vector_type(16))) _Float16 v16h;
typedef __attribute__((ext_vector_type(8)))  float    v8f;

union FragH { v16h v; unsigned int u[8]; };
union FragF { v8f  v; float f[8]; };

static __device__ __forceinline__ float sigmoidf_(float x) {
  return __builtin_amdgcn_rcpf(1.f + __expf(-x));
}
static __device__ __forceinline__ unsigned short f2h(float x) {
  union { _Float16 h; unsigned short s; } c; c.h = (_Float16)x; return c.s;
}
static __device__ __forceinline__ unsigned comb2(unsigned short lo, unsigned short hi) {
  return (unsigned)lo | ((unsigned)hi << 16);
}

// CDNA5 16-bit A-matrix (16x32) per-lane K index for element pair 2v,2v+1.
static __device__ __forceinline__ int kmapA(int v, int lane) {
  return ((v >> 2) * 16) + ((lane & 16) ? 8 : 0) + ((v & 3) << 1);
}
// CDNA5 16-bit B-matrix (32x16): same K order as A (lanes 0-15 K 0..7,16..23; lanes 16-31 K 8..15,24..31).
static __device__ __forceinline__ int kmapB(int v, int lane) {
  return ((v >> 2) * 16) + ((lane & 16) ? 8 : 0) + ((v & 3) << 1);
}

// ---------------------------------------------------------------------------
// fp32 -> fp16 elementwise convert
// ---------------------------------------------------------------------------
__global__ void k_cvt_h(const float* __restrict__ src, unsigned short* __restrict__ dst,
                        size_t n) {
  size_t i = (size_t)blockIdx.x * blockDim.x + threadIdx.x;
  if (i < n) dst[i] = f2h(src[i]);
}

// Weight convert + transpose: WT[n*K+k] = f16(W[k*N+n]); zero-pad n in [N,Npad)
__global__ void k_cvt_wT(const float* __restrict__ W, unsigned short* __restrict__ o,
                         int K, int N, int Npad) {
  int idx = blockIdx.x * 256 + threadIdx.x;
  if (idx >= Npad * K) return;
  int n = idx / K, k = idx - n * K;
  o[idx] = f2h((n < N) ? W[(size_t)k * N + n] : 0.f);
}

// ---------------------------------------------------------------------------
// Generic GEMM on fp16 operands: val = f(Ah[M,K] @ WT^T + bias)
//   Ah: fp16 [M,K] row-major; WT: fp16 [Npad,K] (transposed weights)
//   gate (WT2): val = (A@W+bias) * sigmoid(A@W2+bias2)
//   act: 0 none 1 relu 2 sigmoid ; maskRow per-row scalar ; outScale
//   out (fp32, nullable, addOut fuses residual) ; outH (fp16, nullable)
// Tiles staged with async global->LDS b128 copies, double buffered.
// ---------------------------------------------------------------------------
__global__ __launch_bounds__(256) void k_gemm(
    const unsigned short* __restrict__ Ah, const unsigned short* __restrict__ WT,
    const float* __restrict__ bias, const unsigned short* __restrict__ WT2,
    const float* __restrict__ bias2, const float* __restrict__ maskRow,
    float* __restrict__ out, unsigned short* __restrict__ outH, float outScale,
    int M, int N, int K, int act, int addOut)
{
  __shared__ unsigned short As[2][64][32];
  __shared__ unsigned short Bs[2][64][32];
  __shared__ unsigned short B2s[2][64][32];
  const bool gate = (WT2 != nullptr);
  const int tid = threadIdx.x, lane = tid & 31, wave = tid >> 5;
  const int wm = wave >> 1, wn = wave & 1;
  const int bm = blockIdx.y * 64, bn = blockIdx.x * 64;
  // LDS byte offsets (low 32 bits of generic shared address = LDS offset)
  const unsigned aBase  = (unsigned)(unsigned long long)(const void*)&As[0][0][0];
  const unsigned bBase  = (unsigned)(unsigned long long)(const void*)&Bs[0][0][0];
  const unsigned b2Base = (unsigned)(unsigned long long)(const void*)&B2s[0][0][0];
  const int crow = tid >> 2, cch = (tid & 3) * 8;  // 16 bytes per thread

  FragF acc[2], acc2[2];
  for (int t = 0; t < 2; ++t)
    for (int i = 0; i < 8; ++i) { acc[t].f[i] = 0.f; acc2[t].f[i] = 0.f; }

  auto issue = [&](int buf, int k0) {
    unsigned loff = (unsigned)(((buf * 64 + crow) * 32 + cch) * 2);
    const unsigned short* ga = Ah + (size_t)(bm + crow) * K + k0 + cch;
    asm volatile("global_load_async_to_lds_b128 %0, %1, off"
                 :: "v"(aBase + loff), "v"(ga) : "memory");
    const unsigned short* gb = WT + (size_t)(bn + crow) * K + k0 + cch;
    asm volatile("global_load_async_to_lds_b128 %0, %1, off"
                 :: "v"(bBase + loff), "v"(gb) : "memory");
    if (gate) {
      const unsigned short* g2 = WT2 + (size_t)(bn + crow) * K + k0 + cch;
      asm volatile("global_load_async_to_lds_b128 %0, %1, off"
                   :: "v"(b2Base + loff), "v"(g2) : "memory");
    }
  };

  issue(0, 0);
  const int steps = K / 32;
  for (int s = 0; s < steps; ++s) {
    const int cur = s & 1;
    const bool more = (s + 1 < steps);
    if (more) {
      issue(cur ^ 1, (s + 1) * 32);
      // current tile's copies retired once outstanding <= next tile's count
      if (gate) asm volatile("s_wait_asynccnt 3" ::: "memory");
      else      asm volatile("s_wait_asynccnt 2" ::: "memory");
    } else {
      asm volatile("s_wait_asynccnt 0" ::: "memory");
    }
    __syncthreads();

    FragH fa;
    int arow = wm * 16 + (lane & 15);
    for (int v = 0; v < 8; ++v)
      fa.u[v] = *(const unsigned int*)&As[cur][arow][kmapA(v, lane)];
    for (int nt = 0; nt < 2; ++nt) {
      FragH fb;
      int bcol = wn * 32 + nt * 16 + (lane & 15);
      for (int v = 0; v < 8; ++v)
        fb.u[v] = *(const unsigned int*)&Bs[cur][bcol][kmapB(v, lane)];
      acc[nt].v = __builtin_amdgcn_wmma_f32_16x16x32_f16(
          false, fa.v, false, fb.v, (short)0, acc[nt].v, false, false);
      if (gate) {
        FragH fb2;
        for (int v = 0; v < 8; ++v)
          fb2.u[v] = *(const unsigned int*)&B2s[cur][bcol][kmapB(v, lane)];
        acc2[nt].v = __builtin_amdgcn_wmma_f32_16x16x32_f16(
            false, fa.v, false, fb2.v, (short)0, acc2[nt].v, false, false);
      }
    }
    __syncthreads();
  }

  for (int nt = 0; nt < 2; ++nt) {
    int col = bn + wn * 32 + nt * 16 + (lane & 15);
    if (col >= N) continue;
    for (int r = 0; r < 8; ++r) {
      int row = bm + wm * 16 + r + ((lane & 16) ? 8 : 0);
      float val = acc[nt].f[r] + bias[col];
      if (gate)          val *= sigmoidf_(acc2[nt].f[r] + bias2[col]);
      else if (act == 1) val = fmaxf(val, 0.f);
      else if (act == 2) val = sigmoidf_(val);
      if (maskRow) val *= maskRow[row];
      val *= outScale;
      size_t o = (size_t)row * N + col;
      if (out)  { if (addOut) out[o] += val; else out[o] = val; }
      if (outH) outH[o] = f2h(val);
    }
  }
}

// ---------------------------------------------------------------------------
// LayerNorm over last dim C (one row per block, 128 threads).
// ---------------------------------------------------------------------------
__global__ __launch_bounds__(128) void k_layernorm(
    const float* __restrict__ x, const float* __restrict__ w,
    const float* __restrict__ b, float* __restrict__ out, int C)
{
  const int row = blockIdx.x, tid = threadIdx.x;
  __shared__ float s1[128], s2[128];
  const float* xr = x + (size_t)row * C;
  float s = 0.f, ss = 0.f;
  for (int c = tid; c < C; c += 128) { float v = xr[c]; s += v; ss += v * v; }
  s1[tid] = s; s2[tid] = ss; __syncthreads();
  for (int off = 64; off > 0; off >>= 1) {
    if (tid < off) { s1[tid] += s1[tid + off]; s2[tid] += s2[tid + off]; }
    __syncthreads();
  }
  float mean = s1[0] / C;
  float var  = s2[0] / C - mean * mean;
  float inv  = rsqrtf(var + 1e-5f);
  float* orow = out + (size_t)row * C;
  for (int c = tid; c < C; c += 128) orow[c] = (xr[c] - mean) * inv * w[c] + b[c];
}

// pb[h, p] = zln[p,:] @ w[:,h] + b[h]
__global__ void k_pair_bias(const float* __restrict__ zln,
                            const float* __restrict__ w,
                            const float* __restrict__ b,
                            float* __restrict__ pb, int Npos, int C, int H)
{
  int p = blockIdx.x * blockDim.x + threadIdx.x;
  if (p >= Npos) return;
  const float* zr = zln + (size_t)p * C;
  for (int h = 0; h < H; ++h) {
    float acc = b[h];
    for (int c = 0; c < C; ++c) acc += zr[c] * w[c * H + h];
    pb[(size_t)h * Npos + p] = acc;
  }
}

// ---------------------------------------------------------------------------
// Attention core on fp16 q/k/v (scale pre-folded into q). One wave per
// (16 q rows, head, batch row). Layout [R, N, H*32]. pb [H,Nq,Nk]. mask [R,Nk].
// ---------------------------------------------------------------------------
__global__ __launch_bounds__(32) void k_attention(
    const unsigned short* __restrict__ q_, const unsigned short* __restrict__ k_,
    const unsigned short* __restrict__ v_, const float* __restrict__ pb,
    const float* __restrict__ mask, float* __restrict__ out,
    int Nq, int Nk, int H)
{
  __shared__ float    sc[16][256];
  __shared__ _Float16 pr[16][256];
  const int rr = blockIdx.z, h = blockIdx.y, qt = blockIdx.x;
  const int lane = threadIdx.x;
  const int HC = H * 32;
  const size_t rbase = (size_t)rr * Nq * HC;

  FragH fq;
  {
    int qrow = qt * 16 + (lane & 15);
    const unsigned short* qp = q_ + rbase + (size_t)qrow * HC + h * 32;
    for (int v = 0; v < 8; ++v)
      fq.u[v] = *(const unsigned int*)&qp[kmapA(v, lane)];
  }

  for (int kt = 0; kt < Nk / 16; ++kt) {
    FragH fk;
    int kj = kt * 16 + (lane & 15);
    const unsigned short* kp = k_ + rbase + (size_t)kj * HC + h * 32;
    for (int v = 0; v < 8; ++v)
      fk.u[v] = *(const unsigned int*)&kp[kmapB(v, lane)];
    FragF s0; for (int i = 0; i < 8; ++i) s0.f[i] = 0.f;
    s0.v = __builtin_amdgcn_wmma_f32_16x16x32_f16(
        false, fq.v, false, fk.v, (short)0, s0.v, false, false);
    for (int r = 0; r < 8; ++r) {
      int qr  = r + ((lane & 16) ? 8 : 0);
      int col = kt * 16 + (lane & 15);
      int qg  = qt * 16 + qr;
      sc[qr][col] = s0.f[r]
                  + pb[((size_t)h * Nq + qg) * Nk + col]
                  + (mask[(size_t)rr * Nk + col] - 1.f) * 1e9f;
    }
  }
  __syncthreads();

  for (int row = 0; row < 16; ++row) {
    float mx = -3.4e38f;
    for (int c = lane; c < 256; c += 32) mx = fmaxf(mx, sc[row][c]);
    for (int m = 16; m >= 1; m >>= 1) mx = fmaxf(mx, __shfl_xor(mx, m, 32));
    float sum = 0.f;
    for (int c = lane; c < 256; c += 32) {
      float e = __expf(sc[row][c] - mx);
      sc[row][c] = e; sum += e;
    }
    for (int m = 16; m >= 1; m >>= 1) sum += __shfl_xor(sum, m, 32);
    float inv = __builtin_amdgcn_rcpf(sum);
    for (int c = lane; c < 256; c += 32) pr[row][c] = (_Float16)(sc[row][c] * inv);
  }
  __syncthreads();

  for (int nt = 0; nt < 2; ++nt) {
    FragF oacc; for (int i = 0; i < 8; ++i) oacc.f[i] = 0.f;
    for (int kt = 0; kt < Nk / 32; ++kt) {
      FragH fp;
      int prow = lane & 15;
      for (int v = 0; v < 8; ++v)
        fp.u[v] = *(const unsigned int*)&pr[prow][kt * 32 + kmapA(v, lane)];
      FragH fv;
      int c = nt * 16 + (lane & 15);
      for (int v = 0; v < 8; ++v) {
        int kj = kt * 32 + kmapB(v, lane);
        const unsigned short* vp = v_ + rbase + (size_t)kj * HC + h * 32 + c;
        fv.u[v] = comb2(vp[0], vp[HC]);
      }
      oacc.v = __builtin_amdgcn_wmma_f32_16x16x32_f16(
          false, fp.v, false, fv.v, (short)0, oacc.v, false, false);
    }
    for (int r = 0; r < 8; ++r) {
      int qr  = qt * 16 + r + ((lane & 16) ? 8 : 0);
      int col = h * 32 + nt * 16 + (lane & 15);
      out[rbase + (size_t)qr * HC + col] = oacc.f[r];
    }
  }
}

// ---------------------------------------------------------------------------
// Outer-product-mean (fp16 a,b [128,256,32]; fp16 Wo [1024,128]).
// One wave per (i, 16-wide j tile).
// ---------------------------------------------------------------------------
__global__ __launch_bounds__(32) void k_opm(
    const unsigned short* __restrict__ a, const unsigned short* __restrict__ b,
    const unsigned short* __restrict__ woh, const float* __restrict__ bo,
    const float* __restrict__ mask, float* __restrict__ z)
{
  __shared__ _Float16 outerh[16][1024];
  __shared__ float norm16[16];
  const int i = blockIdx.y, jt = blockIdx.x, lane = threadIdx.x;

  {
    int jl = lane & 15, half = (lane >> 4) & 1;
    float nv = 0.f;
    for (int s = half * 64; s < half * 64 + 64; ++s)
      nv += mask[s * 256 + i] * mask[s * 256 + jt * 16 + jl];
    nv += __shfl_xor(nv, 16, 32);
    if (lane < 16) norm16[lane] = nv;
  }
  __syncthreads();

  for (int jl = 0; jl < 16; ++jl) {
    int j = jt * 16 + jl;
    for (int ct = 0; ct < 2; ++ct)
      for (int dt = 0; dt < 2; ++dt) {
        FragF acc; for (int t = 0; t < 8; ++t) acc.f[t] = 0.f;
        for (int st = 0; st < 4; ++st) {
          FragH fa;
          int c = ct * 16 + (lane & 15);
          for (int v = 0; v < 8; ++v) {
            int s = st * 32 + kmapA(v, lane);
            fa.u[v] = comb2(a[((size_t)s * 256 + i) * 32 + c],
                            a[((size_t)(s + 1) * 256 + i) * 32 + c]);
          }
          FragH fb;
          int d = dt * 16 + (lane & 15);
          for (int v = 0; v < 8; ++v) {
            int s = st * 32 + kmapB(v, lane);
            fb.u[v] = comb2(b[((size_t)s * 256 + j) * 32 + d],
                            b[((size_t)(s + 1) * 256 + j) * 32 + d]);
          }
          acc.v = __builtin_amdgcn_wmma_f32_16x16x32_f16(
              false, fa.v, false, fb.v, (short)0, acc.v, false, false);
        }
        for (int r = 0; r < 8; ++r) {
          int cc = ct * 16 + r + ((lane & 16) ? 8 : 0);
          int dd = dt * 16 + (lane & 15);
          outerh[jl][cc * 32 + dd] = (_Float16)acc.f[r];
        }
      }
  }
  __syncthreads();

  for (int nt = 0; nt < 8; ++nt) {
    FragF acc; for (int t = 0; t < 8; ++t) acc.f[t] = 0.f;
    for (int kt = 0; kt < 32; ++kt) {
      FragH fa;
      int row = lane & 15;
      for (int v = 0; v < 8; ++v)
        fa.u[v] = *(const unsigned int*)&outerh[row][kt * 32 + kmapA(v, lane)];
      FragH fb;
      int e = nt * 16 + (lane & 15);
      for (int v = 0; v < 8; ++v) {
        int kk = kt * 32 + kmapB(v, lane);
        fb.u[v] = comb2(woh[(size_t)kk * 128 + e], woh[(size_t)(kk + 1) * 128 + e]);
      }
      acc.v = __builtin_amdgcn_wmma_f32_16x16x32_f16(
          false, fa.v, false, fb.v, (short)0, acc.v, false, false);
    }
    for (int r = 0; r < 8; ++r) {
      int jl = r + ((lane & 16) ? 8 : 0);
      int j  = jt * 16 + jl;
      int e  = nt * 16 + (lane & 15);
      z[((size_t)i * 256 + j) * 128 + e] +=
          (acc.f[r] + bo[e]) * __builtin_amdgcn_rcpf(1e-3f + norm16[jl]);
    }
  }
}

// ---------------------------------------------------------------------------
// Triangle-mult einsum on fp16 a,b [256,256,128] -> fp32 x.
// ---------------------------------------------------------------------------
__global__ __launch_bounds__(32) void k_trimul(
    const unsigned short* __restrict__ a, const unsigned short* __restrict__ b,
    float* __restrict__ x, int outgoing)
{
  const int c  = blockIdx.z;
  const int i0 = blockIdx.y * 16, j0 = blockIdx.x * 16;
  const int lane = threadIdx.x;
  FragF acc; for (int t = 0; t < 8; ++t) acc.f[t] = 0.f;
  for (int kt = 0; kt < 8; ++kt) {
    FragH fa;
    int i = i0 + (lane & 15);
    for (int v = 0; v < 8; ++v) {
      int kk = kt * 32 + kmapA(v, lane);
      size_t p0 = outgoing ? ((size_t)i * 256 + kk)     : ((size_t)kk * 256 + i);
      size_t p1 = outgoing ? ((size_t)i * 256 + kk + 1) : ((size_t)(kk + 1) * 256 + i);
      fa.u[v] = comb2(a[p0 * 128 + c], a[p1 * 128 + c]);
    }
    FragH fb;
    int j = j0 + (lane & 15);
    for (int v = 0; v < 8; ++v) {
      int kk = kt * 32 + kmapB(v, lane);
      size_t p0 = outgoing ? ((size_t)j * 256 + kk)     : ((size_t)kk * 256 + j);
      size_t p1 = outgoing ? ((size_t)j * 256 + kk + 1) : ((size_t)(kk + 1) * 256 + j);
      fb.u[v] = comb2(b[p0 * 128 + c], b[p1 * 128 + c]);
    }
    acc.v = __builtin_amdgcn_wmma_f32_16x16x32_f16(
        false, fa.v, false, fb.v, (short)0, acc.v, false, false);
  }
  for (int r = 0; r < 8; ++r) {
    int i = i0 + r + ((lane & 16) ? 8 : 0);
    int j = j0 + (lane & 15);
    x[((size_t)i * 256 + j) * 128 + c] = acc.f[r];
  }
}

// --- small helpers ---------------------------------------------------------
__global__ void k_transpose_z(const float* __restrict__ src, float* __restrict__ dst) {
  int i = blockIdx.y, j = blockIdx.x, c = threadIdx.x;
  dst[((size_t)j * 256 + i) * 128 + c] = src[((size_t)i * 256 + j) * 128 + c];
}
__global__ void k_transpose_mask(const float* __restrict__ src, float* __restrict__ dst) {
  int i = blockIdx.x, j = threadIdx.x;
  dst[j * 256 + i] = src[i * 256 + j];
}
__global__ void k_mul(float* __restrict__ y, const float* __restrict__ x, size_t n) {
  size_t i = (size_t)blockIdx.x * blockDim.x + threadIdx.x;
  if (i < n) y[i] *= x[i];
}
__global__ void k_fma_add(float* __restrict__ y, const float* __restrict__ a,
                          const float* __restrict__ b, size_t n) {
  size_t i = (size_t)blockIdx.x * blockDim.x + threadIdx.x;
  if (i < n) y[i] += a[i] * b[i];
}

// ---------------------------------------------------------------------------
// Host orchestration (JAX pytree flatten order: sorted dict keys, b before w).
// ---------------------------------------------------------------------------
extern "C" void kernel_launch(void* const* d_in, const int* in_sizes, int n_in,
                              void* d_out, int out_size, void* d_ws, size_t ws_size,
                              hipStream_t stream) {
  (void)in_sizes; (void)n_in; (void)out_size; (void)ws_size;
  auto IN = [&](int i) { return (const float*)d_in[i]; };

  const float* m_in = IN(0);
  const float* z_in = IN(1);
  const float* msa_mask  = IN(2);
  const float* pair_mask = IN(3);

  const int mar_g_b=4, mar_g_w=5, mar_k_b=6, mar_k_w=7, mar_ln_b=8, mar_ln_w=9,
            mar_lnz_b=10, mar_lnz_w=11, mar_o_b=12, mar_o_w=13, mar_pb_b=14,
            mar_pb_w=15, mar_q_b=16, mar_q_w=17, mar_v_b=18, mar_v_w=19;
  const int mt_ln_b=20, mt_ln_w=21, mt_w1_b=22, mt_w1_w=23, mt_w2_b=24, mt_w2_w=25;
  const int opm_a_b=26, opm_a_w=27, opm_b_b=28, opm_b_w=29, opm_ln_b=30,
            opm_ln_w=31, opm_o_b=32, opm_o_w=33;
  const int pt_ln_b=34, pt_ln_w=35, pt_w1_b=36, pt_w1_w=37, pt_w2_b=38, pt_w2_w=39;
  const int tae0=40, tas0=54, tmi0=68, tmo0=84;

  const size_t U = 8388608;
  float* Mout = (float*)d_out;
  float* Zout = Mout + U;
  float* f  = (float*)d_ws;
  float* B0 = f;        float* B1 = f + U;   float* B2 = f + 2*U;
  float* B3 = f + 3*U;  float* B4 = f + 4*U; float* B5 = f + 5*U;
  float* B6 = f + 6*U;
  float* PB    = f + 7*U;
  float* maskT = PB + 524288;
  unsigned short* AH  = (unsigned short*)(f + 8*U);   // fp16 A staging (<=33.5M halves)
  unsigned short* WH1 = (unsigned short*)(f + 10*U);  // transposed fp16 weights
  unsigned short* WH2 = WH1 + 262144;

  auto ew = [&](size_t n) { return dim3((unsigned)((n + 255) / 256)); };
  auto cvt = [&](const float* s, unsigned short* d, size_t n) {
    k_cvt_h<<<ew(n), 256, 0, stream>>>(s, d, n);
  };
  auto gemm = [&](const unsigned short* Ah, const float* W, const float* bias,
                  const float* W2, const float* b2, const float* mr,
                  float* out, unsigned short* outH, float oscale,
                  int M, int N, int K, int act, int add) {
    int Npad = ((N + 63) / 64) * 64;
    k_cvt_wT<<<ew((size_t)Npad * K), 256, 0, stream>>>(W, WH1, K, N, Npad);
    if (W2) k_cvt_wT<<<ew((size_t)Npad * K), 256, 0, stream>>>(W2, WH2, K, N, Npad);
    dim3 g(Npad / 64, M / 64);
    k_gemm<<<g, 256, 0, stream>>>(Ah, WH1, bias, W2 ? WH2 : nullptr, b2, mr,
                                  out, outH, oscale, M, N, K, act, add);
  };
  const float scale = 0.1767766953f;  // 32^-0.5

  hipMemcpyAsync(Mout, m_in, U * sizeof(float), hipMemcpyDeviceToDevice, stream);
  hipMemcpyAsync(Zout, z_in, U * sizeof(float), hipMemcpyDeviceToDevice, stream);

  // ---- Stage 1: MSA row attention with pair bias --------------------------
  k_layernorm<<<32768, 128, 0, stream>>>(Mout, IN(mar_ln_w), IN(mar_ln_b), B0, 256);
  cvt(B0, AH, U);
  k_layernorm<<<65536, 128, 0, stream>>>(Zout, IN(mar_lnz_w), IN(mar_lnz_b), B5, 128);
  k_pair_bias<<<256, 256, 0, stream>>>(B5, IN(mar_pb_w), IN(mar_pb_b), PB, 65536, 128, 8);
  gemm(AH, IN(mar_q_w), IN(mar_q_b), nullptr, nullptr, nullptr,
       nullptr, (unsigned short*)B1, scale, 32768, 256, 256, 0, 0);
  gemm(AH, IN(mar_k_w), IN(mar_k_b), nullptr, nullptr, nullptr,
       nullptr, (unsigned short*)B2, 1.f, 32768, 256, 256, 0, 0);
  gemm(AH, IN(mar_v_w), IN(mar_v_b), nullptr, nullptr, nullptr,
       nullptr, (unsigned short*)B3, 1.f, 32768, 256, 256, 0, 0);
  gemm(AH, IN(mar_g_w), IN(mar_g_b), nullptr, nullptr, nullptr,
       B4, nullptr, 1.f, 32768, 256, 256, 2, 0);
  k_attention<<<dim3(16, 8, 128), 32, 0, stream>>>(
      (unsigned short*)B1, (unsigned short*)B2, (unsigned short*)B3,
      PB, msa_mask, B5, 256, 256, 8);
  k_mul<<<ew(U), 256, 0, stream>>>(B5, B4, U);
  cvt(B5, AH, U);
  gemm(AH, IN(mar_o_w), IN(mar_o_b), nullptr, nullptr, nullptr,
       Mout, nullptr, 1.f, 32768, 256, 256, 0, 1);

  // ---- Stage 2: MSA transition -------------------------------------------
  k_layernorm<<<32768, 128, 0, stream>>>(Mout, IN(mt_ln_w), IN(mt_ln_b), B0, 256);
  cvt(B0, AH, U);
  gemm(AH, IN(mt_w1_w), IN(mt_w1_b), nullptr, nullptr, nullptr,
       nullptr, (unsigned short*)B1, 1.f, 32768, 1024, 256, 1, 0);
  gemm((unsigned short*)B1, IN(mt_w2_w), IN(mt_w2_b), nullptr, nullptr, msa_mask,
       Mout, nullptr, 1.f, 32768, 256, 1024, 0, 1);

  // ---- Stage 3: Outer product mean ---------------------------------------
  k_layernorm<<<32768, 128, 0, stream>>>(Mout, IN(opm_ln_w), IN(opm_ln_b), B0, 256);
  cvt(B0, AH, U);
  gemm(AH, IN(opm_a_w), IN(opm_a_b), nullptr, nullptr, msa_mask,
       nullptr, (unsigned short*)B5, 1.f, 32768, 32, 256, 0, 0);
  gemm(AH, IN(opm_b_w), IN(opm_b_b), nullptr, nullptr, msa_mask,
       nullptr, (unsigned short*)B6, 1.f, 32768, 32, 256, 0, 0);
  cvt(IN(opm_o_w), WH1, 131072);
  k_opm<<<dim3(16, 256), 32, 0, stream>>>((unsigned short*)B5, (unsigned short*)B6,
                                          WH1, IN(opm_o_b), msa_mask, Zout);

  // ---- Stage 4: Triangle multiplications ---------------------------------
  auto tri_mul = [&](int p0, int outgoing) {
    const int a_g_b=p0+0, a_g_w=p0+1, a_p_b=p0+2, a_p_w=p0+3,
              b_g_b=p0+4, b_g_w=p0+5, b_p_b=p0+6, b_p_w=p0+7,
              g_b=p0+8, g_w=p0+9, lnin_b=p0+10, lnin_w=p0+11,
              lnout_b=p0+12, lnout_w=p0+13, z_b=p0+14, z_w=p0+15;
    k_layernorm<<<65536, 128, 0, stream>>>(Zout, IN(lnin_w), IN(lnin_b), B0, 128);
    cvt(B0, AH, U);
    gemm(AH, IN(a_p_w), IN(a_p_b), IN(a_g_w), IN(a_g_b), pair_mask,
         nullptr, (unsigned short*)B1, 1.f, 65536, 128, 128, 0, 0);
    gemm(AH, IN(b_p_w), IN(b_p_b), IN(b_g_w), IN(b_g_b), pair_mask,
         nullptr, (unsigned short*)B2, 1.f, 65536, 128, 128, 0, 0);
    gemm(AH, IN(g_w), IN(g_b), nullptr, nullptr, nullptr,
         B3, nullptr, 1.f, 65536, 128, 128, 2, 0);
    k_trimul<<<dim3(16, 16, 128), 32, 0, stream>>>(
        (unsigned short*)B1, (unsigned short*)B2, B4, outgoing);
    k_layernorm<<<65536, 128, 0, stream>>>(B4, IN(lnout_w), IN(lnout_b), B5, 128);
    cvt(B5, AH, U);
    gemm(AH, IN(z_w), IN(z_b), nullptr, nullptr, nullptr,
         B4, nullptr, 1.f, 65536, 128, 128, 0, 0);
    k_fma_add<<<ew(U), 256, 0, stream>>>(Zout, B4, B3, U);
  };
  tri_mul(tmo0, 1);
  tri_mul(tmi0, 0);

  // ---- Stage 5/6: Triangle attention start / end -------------------------
  auto tri_att = [&](int p0, float* zbuf, const float* mk) {
    const int g_b=p0+0, g_w=p0+1, k_b=p0+2, k_w=p0+3, ln_b=p0+4, ln_w=p0+5,
              o_b=p0+6, o_w=p0+7, pb_b=p0+8, pb_w=p0+9, q_b=p0+10, q_w=p0+11,
              v_b=p0+12, v_w=p0+13;
    k_layernorm<<<65536, 128, 0, stream>>>(zbuf, IN(ln_w), IN(ln_b), B0, 128);
    k_pair_bias<<<256, 256, 0, stream>>>(B0, IN(pb_w), IN(pb_b), PB, 65536, 128, 4);
    cvt(B0, AH, U);
    gemm(AH, IN(q_w), IN(q_b), nullptr, nullptr, nullptr,
         nullptr, (unsigned short*)B1, scale, 65536, 128, 128, 0, 0);
    gemm(AH, IN(k_w), IN(k_b), nullptr, nullptr, nullptr,
         nullptr, (unsigned short*)B2, 1.f, 65536, 128, 128, 0, 0);
    gemm(AH, IN(v_w), IN(v_b), nullptr, nullptr, nullptr,
         nullptr, (unsigned short*)B3, 1.f, 65536, 128, 128, 0, 0);
    gemm(AH, IN(g_w), IN(g_b), nullptr, nullptr, nullptr,
         B4, nullptr, 1.f, 65536, 128, 128, 2, 0);
    k_attention<<<dim3(16, 4, 256), 32, 0, stream>>>(
        (unsigned short*)B1, (unsigned short*)B2, (unsigned short*)B3,
        PB, mk, B5, 256, 256, 4);
    k_mul<<<ew(U), 256, 0, stream>>>(B5, B4, U);
    cvt(B5, AH, U);
    gemm(AH, IN(o_w), IN(o_b), nullptr, nullptr, nullptr,
         zbuf, nullptr, 1.f, 65536, 128, 128, 0, 1);
  };
  tri_att(tas0, Zout, pair_mask);
  k_transpose_z<<<dim3(256, 256), 128, 0, stream>>>(Zout, B6);
  k_transpose_mask<<<256, 256, 0, stream>>>(pair_mask, maskT);
  tri_att(tae0, B6, maskT);
  k_transpose_z<<<dim3(256, 256), 128, 0, stream>>>(B6, Zout);

  // ---- Stage 7: Pair transition ------------------------------------------
  k_layernorm<<<65536, 128, 0, stream>>>(Zout, IN(pt_ln_w), IN(pt_ln_b), B0, 128);
  cvt(B0, AH, U);
  gemm(AH, IN(pt_w1_w), IN(pt_w1_b), nullptr, nullptr, nullptr,
       nullptr, (unsigned short*)B1, 1.f, 65536, 512, 128, 1, 0);
  gemm((unsigned short*)B1, IN(pt_w2_w), IN(pt_w2_b), nullptr, nullptr, pair_mask,
       Zout, nullptr, 1.f, 65536, 128, 512, 0, 1);
}
